// BiLSTM_50500225466406
// MI455X (gfx1250) — matrix operand order repaired and checked by reading, not compiled, not run on hardware
//
#include <hip/hip_runtime.h>
#include <hip/hip_bf16.h>

typedef __attribute__((ext_vector_type(16))) _Float16 v16h;
typedef __attribute__((ext_vector_type(8)))  float    v8f;
typedef __attribute__((ext_vector_type(4)))  unsigned int u32x4;
typedef __attribute__((ext_vector_type(8)))  int i32x8;
typedef __attribute__((ext_vector_type(4)))  int i32x4;

// ---------------------------------------------------------------------------
// Problem constants (from reference): B=32, S=128, E=300, H=200, T=17
// ---------------------------------------------------------------------------
#define BATCH   32
#define SEQ     128
#define EMB     300
#define EMB_P   320     // padded to 32
#define HID     200
#define HID_P   224     // padded to 32
#define GATES   800     // 4*H
#define L1IN    400     // 2*H
#define L1IN_P  416     // padded to 32
#define NTAGS   17
#define ROWS    (SEQ*BATCH)      // 4096 time-major rows (s,b)
#define SLAB    (BATCH*GATES)    // 25600 f32 per timestep per direction

#if __has_builtin(__builtin_amdgcn_tensor_load_to_lds) && __has_builtin(__builtin_amdgcn_s_wait_tensorcnt)
#define HAVE_TDM 1
#else
#define HAVE_TDM 0
#endif

// ---------------------------------------------------------------------------
// WMMA tile loaders (wave32), contiguous k order k = 16*(lane/16) + s for A and B
// ---------------------------------------------------------------------------
// A-matrix 16x32 f16: lane m=L&15; lanes 0-15 hold K=0..15, lanes 16-31 hold K=16..31
__device__ __forceinline__ v16h load_a_tile(const _Float16* tile, int ld, int lane) {
    int m   = lane & 15;
    int khi = (lane & 16) ? 16 : 0;
    const _Float16* p = tile + (size_t)m * ld + khi;
    v16h a;
#pragma unroll
    for (int j = 0; j < 8; ++j) a[j]     = p[j];
#pragma unroll
    for (int j = 0; j < 8; ++j) a[8 + j] = p[8 + j];
    return a;
}

// B-matrix 32x16 f16, element (k,n) = W[n][k] (W row-major N x K):
// lane n=L&15; lanes 0-15 hold K=0..15, lanes 16-31 hold K=16..31
__device__ __forceinline__ v16h load_b_tile(const _Float16* wtile, int ld, int lane) {
    int n  = lane & 15;
    int k0 = (lane & 16) ? 16 : 0;
    const _Float16* p = wtile + (size_t)n * ld + k0;
    v16h b;
#pragma unroll
    for (int j = 0; j < 16; ++j) b[j] = p[j];
    return b;
}

__device__ __forceinline__ float sigf(float x) { return 1.0f / (1.0f + __expf(-x)); }

// ---------------------------------------------------------------------------
// Tensor Data Mover: async DMA one timestep slab (25600 f32) global -> LDS.
// D# packing per CDNA5 ISA §8.3 (group0) / §8.4 (group1):
//   g0: count=1 | lds_addr | global_addr[56:0] | type=2
//   g1: data_size=4B, tensor_dim0=tile_dim0=SLAB, dims1..=1, stride0=SLAB
// This toolchain exposes the 6-arg builtin: (g0, g1, g2, g3, g4, cpol).
// ---------------------------------------------------------------------------
__device__ __forceinline__ void tdm_issue_slab(float* lds_dst, const float* gsrc) {
#if HAVE_TDM
    unsigned long long ga = (unsigned long long)(uintptr_t)gsrc;
    unsigned lds_off = (unsigned)(uintptr_t)(void*)lds_dst;  // flat->LDS: addr[31:0]
    u32x4 g0;
    g0[0] = 1u;                                              // count=1, user mode
    g0[1] = lds_off;                                         // lds_addr (bytes)
    g0[2] = (unsigned)(ga & 0xFFFFFFFFu);                    // global_addr[31:0]
    g0[3] = (unsigned)((ga >> 32) & 0x01FFFFFFu) | 0x80000000u; // addr[56:32] | type=2
    i32x8 g1;
    g1[0] = (int)(2u << 16);                                 // data_size = 4 bytes
    g1[1] = (int)((SLAB & 0xFFFFu) << 16);                   // tensor_dim0[15:0]
    g1[2] = (int)(((SLAB >> 16) & 0xFFFFu) | (1u << 16));    // tensor_dim0[31:16], tensor_dim1=1
    g1[3] = (int)((SLAB & 0xFFFFu) << 16);                   // tile_dim0 = SLAB (16-bit ok: 25600)
    g1[4] = 1;                                               // tile_dim1 = 1
    g1[5] = (int)SLAB;                                       // tensor_dim0_stride lo
    g1[6] = 0;
    g1[7] = 0;
    i32x4 gz4 = {0, 0, 0, 0};
    i32x8 gz8 = {0, 0, 0, 0, 0, 0, 0, 0};
    __builtin_amdgcn_tensor_load_to_lds(g0, g1, gz4, gz4, gz8, 0);
#else
    (void)lds_dst; (void)gsrc;
#endif
}

// ---------------------------------------------------------------------------
// 1) Embedding gather -> f16, K padded. Rows ordered time-major (s,b).
// ---------------------------------------------------------------------------
__global__ void embed_gather_kernel(const int* __restrict__ words,
                                    const float* __restrict__ emb,
                                    _Float16* __restrict__ x_f16) {
    int row = blockIdx.x;             // row = s*BATCH + b
    int s = row / BATCH, b = row % BATCH;
    int w = words[b * SEQ + s];       // words is (B,S)
    const float* src = emb + (size_t)w * EMB;
    for (int k = threadIdx.x; k < EMB_P; k += blockDim.x)
        x_f16[(size_t)row * EMB_P + k] = (k < EMB) ? (_Float16)src[k] : (_Float16)0.0f;
}

// ---------------------------------------------------------------------------
// 2) Generic f32 -> f16 convert with K padding (weights + layer-1 input)
// ---------------------------------------------------------------------------
__global__ void pad_convert_kernel(const float* __restrict__ in, _Float16* __restrict__ out,
                                   int K, int Kp) {
    int row = blockIdx.x;
    for (int k = threadIdx.x; k < Kp; k += blockDim.x)
        out[(size_t)row * Kp + k] = (k < K) ? (_Float16)in[(size_t)row * K + k] : (_Float16)0.0f;
}

// ---------------------------------------------------------------------------
// 3) Input-projection GEMM: out(ROWS x GATES) = X(ROWS x Kp) @ W(GATES x Kp)^T
//    One wave computes a 16x32 strip (2 N-tiles) per job.
// ---------------------------------------------------------------------------
__global__ void gemm_wmma_kernel(const _Float16* __restrict__ X,
                                 const _Float16* __restrict__ W,
                                 float* __restrict__ out, int Kp) {
    const int lane   = threadIdx.x & 31;
    const int waveIB = threadIdx.x >> 5;              // 8 waves / block
    const int job    = blockIdx.x * 8 + waveIB;       // jobs = 256 * 25 = 6400
    const int mTile  = job / (GATES / 32);            // 0..255
    const int nGrp   = job % (GATES / 32);            // 0..24
    const int m0 = mTile * 16, n0 = nGrp * 32;

    v8f acc0 = {}, acc1 = {};
    const _Float16* xTile = X + (size_t)m0 * Kp;
    for (int k0 = 0; k0 < Kp; k0 += 32) {
        v16h a  = load_a_tile(xTile + k0, Kp, lane);
        v16h b0 = load_b_tile(W + (size_t)n0 * Kp + k0, Kp, lane);
        v16h b1 = load_b_tile(W + (size_t)(n0 + 16) * Kp + k0, Kp, lane);
        acc0 = __builtin_amdgcn_wmma_f32_16x16x32_f16(false, a, false, b0, (short)0, acc0, false, false);
        acc1 = __builtin_amdgcn_wmma_f32_16x16x32_f16(false, a, false, b1, (short)0, acc1, false, false);
    }
    // C/D layout: M = r + 8*(lane>=16), N = lane&15
    const int n = lane & 15, mh = (lane & 16) ? 8 : 0;
#pragma unroll
    for (int r = 0; r < 8; ++r) {
        int m = m0 + r + mh;
        out[(size_t)m * GATES + n0 + n]      = acc0[r];
        out[(size_t)m * GATES + n0 + 16 + n] = acc1[r];
    }
}

// ---------------------------------------------------------------------------
// 4) Persistent recurrent kernel: one 32-wave block per direction.
//    Per step: gates(32x800) = preact[t] + bias + h @ Whh^T  (WMMA, K=224)
//    preact slab is TDM-prefetched into a double-buffered LDS slab one step
//    ahead; gates overwrite the consumed slab in place. h is LDS f16, c LDS f32.
// ---------------------------------------------------------------------------
__global__ void lstm_recurrent_kernel(const float* __restrict__ preact,   // (2, ROWS, GATES)
                                      const _Float16* __restrict__ Whh,   // (2, GATES, HID_P)
                                      const float* __restrict__ bias,     // (2, GATES)
                                      float* __restrict__ hout,           // (ROWS, 2*HID)
                                      int outStride, int outOff) {
    extern __shared__ char smem[];
    _Float16* sh_h = (_Float16*)smem;                                     // 32 x HID_P  f16
    float*    sh_c = (float*)(smem + BATCH * HID_P * sizeof(_Float16));   // 32 x HID    f32
    float*    slab0 = sh_c + BATCH * HID;                                 // 32 x GATES  f32
    float*    slab1 = slab0 + SLAB;                                       // 32 x GATES  f32

    const int dir  = blockIdx.x;
    const int tid  = threadIdx.x;
    const int lane = tid & 31;
    const int wave = tid >> 5;            // 0..31

    const float*    pre_d = preact + (size_t)dir * ROWS * GATES;
    const _Float16* W_d   = Whh + (size_t)dir * GATES * HID_P;
    const float*    b_d   = bias + (size_t)dir * GATES;

    // zero-init state
    for (int i = tid; i < BATCH * HID_P; i += blockDim.x) sh_h[i] = (_Float16)0.0f;
    for (int i = tid; i < BATCH * HID;  i += blockDim.x) sh_c[i] = 0.0f;
    __syncthreads();

    // prologue: prefetch slab for step 0
    {
        const int pos0 = (dir == 0) ? 0 : (SEQ - 1);
#if HAVE_TDM
        if (wave == 0) tdm_issue_slab(slab0, pre_d + (size_t)pos0 * SLAB);
#else
        for (int i = tid; i < SLAB; i += blockDim.x)
            slab0[i] = pre_d[(size_t)pos0 * SLAB + i];
#endif
    }

    for (int step = 0; step < SEQ; ++step) {
        const int pos = (dir == 0) ? step : (SEQ - 1 - step);
        float* cur  = (step & 1) ? slab1 : slab0;
        float* next = (step & 1) ? slab0 : slab1;

        // prefetch next step's slab into the other buffer (its old gates were
        // fully consumed before the previous iteration's trailing barrier)
        if (step + 1 < SEQ) {
            const int posN = (dir == 0) ? (step + 1) : (SEQ - 2 - step);
#if HAVE_TDM
            if (wave == 0) tdm_issue_slab(next, pre_d + (size_t)posN * SLAB);
#else
            for (int i = tid; i < SLAB; i += blockDim.x)
                next[i] = pre_d[(size_t)posN * SLAB + i];
#endif
        }
#if HAVE_TDM
        // in-order TDM completion: cnt<=1 means the older (current) slab landed
        if (step + 1 < SEQ) __builtin_amdgcn_s_wait_tensorcnt(1);
        else                __builtin_amdgcn_s_wait_tensorcnt(0);
#endif
        __syncthreads();   // publish current slab to all waves

        // --- gates = slab(preact) + bias + h @ Whh^T, written back in place --
        // 100 output tiles (2 M-tiles x 50 N-tiles) over 32 waves
        for (int tile = wave; tile < (BATCH / 16) * (GATES / 16); tile += 32) {
            const int m0 = (tile & 1) * 16;
            const int n0 = (tile >> 1) * 16;
            const int n  = lane & 15, mh = (lane & 16) ? 8 : 0;

            v8f acc;
#pragma unroll
            for (int r = 0; r < 8; ++r) {
                int m = m0 + r + mh;
                acc[r] = cur[m * GATES + n0 + n] + b_d[n0 + n];
            }
#pragma unroll
            for (int k0 = 0; k0 < HID_P; k0 += 32) {
                v16h a = load_a_tile(sh_h + (size_t)m0 * HID_P + k0, HID_P, lane);
                v16h b = load_b_tile(W_d + (size_t)n0 * HID_P + k0, HID_P, lane);
                acc = __builtin_amdgcn_wmma_f32_16x16x32_f16(false, a, false, b, (short)0, acc, false, false);
            }
#pragma unroll
            for (int r = 0; r < 8; ++r) {
                int m = m0 + r + mh;
                cur[m * GATES + n0 + n] = acc[r];
            }
        }
        __syncthreads();

        // --- elementwise LSTM cell update -------------------------------
        for (int idx = tid; idx < BATCH * HID; idx += blockDim.x) {
            const int b = idx / HID, n = idx % HID;
            const float gi = cur[b * GATES + n];
            const float gf = cur[b * GATES + HID + n];
            const float gg = cur[b * GATES + 2 * HID + n];
            const float go = cur[b * GATES + 3 * HID + n];
            float c = sigf(gf) * sh_c[idx] + sigf(gi) * tanhf(gg);
            float h = sigf(go) * tanhf(c);
            sh_c[idx] = c;
            sh_h[b * HID_P + n] = (_Float16)h;
            hout[((size_t)pos * BATCH + b) * outStride + outOff + dir * HID + n] = h;
        }
        __syncthreads();
    }
}

// ---------------------------------------------------------------------------
// 5) Output head: sigmoid(h1out @ out_w^T + out_b); out is (B, S, 17)
// ---------------------------------------------------------------------------
__global__ void output_kernel(const float* __restrict__ h1,    // (ROWS, 400) time-major
                              const float* __restrict__ out_w, // (17, 400)
                              const float* __restrict__ out_b,
                              float* __restrict__ out) {
    int idx = blockIdx.x * blockDim.x + threadIdx.x;
    if (idx >= BATCH * SEQ * NTAGS) return;
    const int t = idx % NTAGS;
    const int s = (idx / NTAGS) % SEQ;
    const int b = idx / (NTAGS * SEQ);
    const float* hrow = h1 + ((size_t)s * BATCH + b) * L1IN;
    const float* wrow = out_w + (size_t)t * L1IN;
    float acc = out_b[t];
#pragma unroll 4
    for (int k = 0; k < L1IN; ++k) acc += hrow[k] * wrow[k];
    out[idx] = sigf(acc);
}

// ---------------------------------------------------------------------------
// Host-side orchestration
// ---------------------------------------------------------------------------
extern "C" void kernel_launch(void* const* d_in, const int* in_sizes, int n_in,
                              void* d_out, int out_size, void* d_ws, size_t ws_size,
                              hipStream_t stream) {
    (void)in_sizes; (void)n_in; (void)out_size; (void)ws_size;

    const int*   words     = (const int*)  d_in[0];
    const float* emb_table = (const float*)d_in[3];
    const float* Wih0      = (const float*)d_in[7];   // (2,800,300)
    const float* Whh0      = (const float*)d_in[8];   // (2,800,200)
    const float* b0        = (const float*)d_in[9];   // (2,800)
    const float* Wih1      = (const float*)d_in[10];  // (2,800,400)
    const float* Whh1      = (const float*)d_in[11];  // (2,800,200)
    const float* b1        = (const float*)d_in[12];  // (2,800)
    const float* out_w     = (const float*)d_in[13];  // (17,400)
    const float* out_b     = (const float*)d_in[14];  // (17,)
    float*       out       = (float*)d_out;

    // ---- workspace carve-up (all 256B aligned) ----
    char* ws = (char*)d_ws;
    size_t off = 0;
    auto carve = [&](size_t bytes) { char* p = ws + off; off += (bytes + 255) & ~size_t(255); return p; };

    _Float16* x0_f16   = (_Float16*)carve((size_t)ROWS * EMB_P * 2);       // 2.6 MB
    _Float16* Wih0_f16 = (_Float16*)carve((size_t)2 * GATES * EMB_P * 2);  // 1.0 MB
    _Float16* Whh0_f16 = (_Float16*)carve((size_t)2 * GATES * HID_P * 2);  // 0.7 MB
    _Float16* Wih1_f16 = (_Float16*)carve((size_t)2 * GATES * L1IN_P * 2); // 1.3 MB
    _Float16* Whh1_f16 = (_Float16*)carve((size_t)2 * GATES * HID_P * 2);  // 0.7 MB
    float*    preact   = (float*)   carve((size_t)2 * ROWS * GATES * 4);   // 26.2 MB (reused L0/L1)
    float*    h0out    = (float*)   carve((size_t)ROWS * L1IN * 4);        // 6.6 MB
    _Float16* x1_f16   = (_Float16*)carve((size_t)ROWS * L1IN_P * 2);      // 3.4 MB
    float*    h1out    = (float*)   carve((size_t)ROWS * L1IN * 4);        // 6.6 MB

    const size_t lstmShm = BATCH * HID_P * sizeof(_Float16)   // h (f16)
                         + BATCH * HID  * sizeof(float)       // c
                         + (size_t)2 * SLAB * sizeof(float);  // double-buffered pre/gates

    // 1) embedding gather (time-major rows), weight conversions
    embed_gather_kernel<<<ROWS, 128, 0, stream>>>(words, emb_table, x0_f16);
    pad_convert_kernel<<<2 * GATES, 128, 0, stream>>>(Wih0, Wih0_f16, EMB,  EMB_P);
    pad_convert_kernel<<<2 * GATES, 128, 0, stream>>>(Whh0, Whh0_f16, HID,  HID_P);
    pad_convert_kernel<<<2 * GATES, 128, 0, stream>>>(Wih1, Wih1_f16, L1IN, L1IN_P);
    pad_convert_kernel<<<2 * GATES, 128, 0, stream>>>(Whh1, Whh1_f16, HID,  HID_P);

    // 2) layer-0 input projections (both directions), then recurrence
    const int gemmGrid = (ROWS / 16) * (GATES / 32) / 8;   // 6400 wave-jobs / 8 waves
    for (int dir = 0; dir < 2; ++dir)
        gemm_wmma_kernel<<<gemmGrid, 256, 0, stream>>>(
            x0_f16, Wih0_f16 + (size_t)dir * GATES * EMB_P,
            preact + (size_t)dir * ROWS * GATES, EMB_P);
    lstm_recurrent_kernel<<<2, 1024, lstmShm, stream>>>(preact, Whh0_f16, b0, h0out, L1IN, 0);

    // 3) layer-1: convert h0out to padded f16, input projections, recurrence
    pad_convert_kernel<<<ROWS, 256, 0, stream>>>(h0out, x1_f16, L1IN, L1IN_P);
    for (int dir = 0; dir < 2; ++dir)
        gemm_wmma_kernel<<<gemmGrid, 256, 0, stream>>>(
            x1_f16, Wih1_f16 + (size_t)dir * GATES * L1IN_P,
            preact + (size_t)dir * ROWS * GATES, L1IN_P);
    lstm_recurrent_kernel<<<2, 1024, lstmShm, stream>>>(preact, Whh1_f16, b1, h1out, L1IN, 0);

    // 4) output head
    const int nOut = BATCH * SEQ * NTAGS;
    output_kernel<<<(nOut + 255) / 256, 256, 0, stream>>>(h1out, out_w, out_b, out);
}
